// LHATransformerBlock_51479478010637
// MI455X (gfx1250) — matrix operand order repaired and checked by reading, not compiled, not run on hardware
//
#include <hip/hip_runtime.h>
#include <cstdint>
#include <cstddef>

// ---------------- problem dims (fixed by the reference) ----------------
#define BB   2
#define SS   2048
#define DDIM 1024
#define HH   16
#define DHD  64
#define NBK  32
#define MLPD 4096
#define BSR  (BB * SS)   // 4096 rows
#define KFD  96          // packed attention feature dim: 64 content + 32 bucket

typedef __attribute__((ext_vector_type(16))) __bf16 bf16x16;
typedef __attribute__((ext_vector_type(8)))  __bf16 bf16x8;
typedef __attribute__((ext_vector_type(8)))  float  f32x8;
typedef __attribute__((ext_vector_type(4)))  int    i32x4;

// ---------------- CDNA5 async global->LDS path (guarded) ----------------
#if defined(__HIP_DEVICE_COMPILE__) && defined(__gfx1250__) && \
    __has_builtin(__builtin_amdgcn_global_load_async_to_lds_b128) && \
    __has_builtin(__builtin_amdgcn_s_wait_asynccnt)
#define HAS_ASYNC 1
#else
#define HAS_ASYNC 0
#endif

__device__ __forceinline__ void async_cp16(const void* g, void* l) {
#if HAS_ASYNC
  auto gp = (__attribute__((address_space(1))) i32x4*)(uintptr_t)g;
  auto lp = (__attribute__((address_space(3))) i32x4*)l;
  __builtin_amdgcn_global_load_async_to_lds_b128(gp, lp, 0, 0);
#else
  *(int4*)l = *(const int4*)g;
#endif
}
__device__ __forceinline__ void async_wait() {
#if HAS_ASYNC
  __builtin_amdgcn_s_wait_asynccnt(0);
#endif
}

// ---------------- WMMA fragment loaders (ISA 7.12.2 layouts) ----------------
// A 16x32 bf16: lanes 0-15 row M=lane, K={0..7,16..23}; lanes 16-31 K={8..15,24..31}
__device__ __forceinline__ bf16x16 load_frag_a(const __bf16* base, int ld) {
  int lane = threadIdx.x & 31;
  int lo = lane & 15, hi = lane >> 4;
  const __bf16* p = base + lo * ld + hi * 8;
  bf16x8 x = *(const bf16x8*)p;
  bf16x8 y = *(const bf16x8*)(p + 16);
  return __builtin_shufflevector(x, y, 0,1,2,3,4,5,6,7,8,9,10,11,12,13,14,15);
}
// B 32x16 bf16 from LDS holding B^T rows: lanes 0-15 col N=lane, K={0..7,16..23}; lanes 16-31 K={8..15,24..31}
__device__ __forceinline__ bf16x16 load_frag_b(const __bf16* base, int ld) {
  int lane = threadIdx.x & 31;
  int lo = lane & 15, hi = lane >> 4;
  const __bf16* p = base + lo * ld + hi * 8;
  bf16x8 x = *(const bf16x8*)p;
  bf16x8 y = *(const bf16x8*)(p + 16);
  return __builtin_shufflevector(x, y, 0,1,2,3,4,5,6,7,8,9,10,11,12,13,14,15);
}

// ---------------- weight convert + transpose: out[n*K+k] = scale*in[k*N+n] ----------------
__global__ void k_convert_t(const float* __restrict__ in, __bf16* __restrict__ out,
                            int K, int N, float scale) {
  int idx = blockIdx.x * 256 + threadIdx.x;
  if (idx >= K * N) return;
  int n = idx / K, k = idx - n * K;
  out[idx] = (__bf16)(in[(size_t)k * N + n] * scale);
}

__global__ void k_zero(float* __restrict__ p, int n) {
  int i = blockIdx.x * 256 + threadIdx.x;
  if (i < n) p[i] = 0.f;
}

// ---------------- LayerNorm (fp32 in, bf16 out), one row (D=1024) per block ----------------
__global__ __launch_bounds__(256) void k_layernorm(const float* __restrict__ x,
                                                   const float* __restrict__ g,
                                                   const float* __restrict__ b,
                                                   __bf16* __restrict__ out) {
  __shared__ float red[2][8];
  int row = blockIdx.x;
  const float* xr = x + (size_t)row * DDIM;
  float v[4], s0 = 0.f, s1 = 0.f;
#pragma unroll
  for (int i = 0; i < 4; i++) {
    v[i] = xr[threadIdx.x + i * 256];
    s0 += v[i]; s1 += v[i] * v[i];
  }
#pragma unroll
  for (int m = 16; m; m >>= 1) { s0 += __shfl_xor(s0, m); s1 += __shfl_xor(s1, m); }
  int w = threadIdx.x >> 5;
  if ((threadIdx.x & 31) == 0) { red[0][w] = s0; red[1][w] = s1; }
  __syncthreads();
  s0 = red[0][threadIdx.x & 7]; s1 = red[1][threadIdx.x & 7];
#pragma unroll
  for (int m = 4; m; m >>= 1) { s0 += __shfl_xor(s0, m); s1 += __shfl_xor(s1, m); }
  float mu = s0 / DDIM;
  float var = s1 / DDIM - mu * mu;
  float rs = rsqrtf(var + 1e-6f);
#pragma unroll
  for (int i = 0; i < 4; i++) {
    int c = threadIdx.x + i * 256;
    out[(size_t)row * DDIM + c] = (__bf16)((v[i] - mu) * rs * g[c] + b[c]);
  }
}

// ---------------- generic WMMA GEMM: C = A(MxK) * Bt(NxK)^T, 128x128 tile ----------------
template <bool OUTB, bool RELU, bool BIAS, bool RES>
__global__ __launch_bounds__(256) void k_gemm(const __bf16* __restrict__ A,
                                              const __bf16* __restrict__ Bt,
                                              int M, int N, int K,
                                              const float* __restrict__ bias,
                                              const float* __restrict__ res,
                                              float* __restrict__ Cf,
                                              __bf16* __restrict__ Cb) {
  constexpr int LDT = 40;  // 32 + pad, 16B aligned rows
  __shared__ __bf16 sA[128 * LDT];
  __shared__ __bf16 sB[128 * LDT];
  const int m0 = blockIdx.y * 128, n0 = blockIdx.x * 128;
  const int t = threadIdx.x, lane = t & 31, w = t >> 5;
  const int wm = (w >> 2) * 64, wn = (w & 3) * 32;
  f32x8 acc[4][2] = {};
  const int lrow = t >> 1, lcol = (t & 1) * 16;

  for (int k0 = 0; k0 < K; k0 += 32) {
    __syncthreads();  // previous-iter LDS reads done before overwrite
    const __bf16* ga = A  + (size_t)(m0 + lrow) * K + k0 + lcol;
    const __bf16* gb = Bt + (size_t)(n0 + lrow) * K + k0 + lcol;
    __bf16* la = sA + lrow * LDT + lcol;
    __bf16* lb = sB + lrow * LDT + lcol;
    async_cp16(ga, la); async_cp16(ga + 8, la + 8);
    async_cp16(gb, lb); async_cp16(gb + 8, lb + 8);
    async_wait();
    __syncthreads();
    bf16x16 bfr[2];
#pragma unroll
    for (int in = 0; in < 2; ++in)
      bfr[in] = load_frag_b(sB + (wn + in * 16) * LDT, LDT);
#pragma unroll
    for (int im = 0; im < 4; ++im) {
      bf16x16 afr = load_frag_a(sA + (wm + im * 16) * LDT, LDT);
#pragma unroll
      for (int in = 0; in < 2; ++in)
        acc[im][in] = __builtin_amdgcn_wmma_f32_16x16x32_bf16(
            false, afr, false, bfr[in], (short)0, acc[im][in], false, false);
    }
  }

  const int lo = lane & 15, hi = lane >> 4;
#pragma unroll
  for (int im = 0; im < 4; ++im)
#pragma unroll
    for (int in = 0; in < 2; ++in) {
      int col = n0 + wn + in * 16 + lo;
      float bv = BIAS ? bias[col] : 0.f;
#pragma unroll
      for (int r = 0; r < 8; ++r) {
        int row = m0 + wm + im * 16 + r + hi * 8;
        float xv = acc[im][in][r] + bv;
        if (RES) xv += res[(size_t)row * N + col];
        if (RELU) xv = fmaxf(xv, 0.f);
        if (OUTB) Cb[(size_t)row * N + col] = (__bf16)xv;
        else      Cf[(size_t)row * N + col] = xv;
      }
    }
}

// ---------------- pack: bucket softmax + feature concat + V transpose ----------------
// one wave per (b,s,h); lane = bucket index n (NB==32==wave32)
__global__ __launch_bounds__(256) void k_pack(const __bf16* __restrict__ q,
                                              const __bf16* __restrict__ k,
                                              const __bf16* __restrict__ v,
                                              const float* __restrict__ Whq,
                                              const float* __restrict__ Whk,
                                              __bf16* __restrict__ Qf,
                                              __bf16* __restrict__ Kf,
                                              __bf16* __restrict__ vT,
                                              float* __restrict__ mq_sum,
                                              float* __restrict__ mk_sum) {
  int widx = blockIdx.x * 8 + (threadIdx.x >> 5);
  int lane = threadIdx.x & 31;
  int h = widx & (HH - 1);
  int bs = widx >> 4;              // b*S + s
  int b = bs >> 11;                // S == 2048
  int s = bs & (SS - 1);
  const __bf16* qr = q + ((size_t)bs * HH + h) * DHD;
  const __bf16* kr = k + ((size_t)bs * HH + h) * DHD;
  const __bf16* vr = v + ((size_t)bs * HH + h) * DHD;
  float q0 = (float)qr[2 * lane], q1 = (float)qr[2 * lane + 1];
  float k0 = (float)kr[2 * lane], k1 = (float)kr[2 * lane + 1];
  // bucket logits (lane = bucket n); note q already carries 1/sqrt(DH) from Wq
  const float* WqH = Whq + (size_t)h * DHD * NBK;
  const float* WkH = Whk + (size_t)h * DHD * NBK;
  float lq = 0.f, lk = 0.f;
  for (int i = 0; i < 32; i++) {
    float a0 = __shfl(q0, i), a1 = __shfl(q1, i);
    float c0 = __shfl(k0, i), c1 = __shfl(k1, i);
    // undo the 1/8 scale for the bucket path (q was pre-scaled): 8x
    lq += 8.f * (a0 * WqH[(2 * i) * NBK + lane] + a1 * WqH[(2 * i + 1) * NBK + lane]);
    lk += c0 * WkH[(2 * i) * NBK + lane] + c1 * WkH[(2 * i + 1) * NBK + lane];
  }
  float mq = lq, mk = lk;
#pragma unroll
  for (int m = 16; m; m >>= 1) { mq = fmaxf(mq, __shfl_xor(mq, m)); mk = fmaxf(mk, __shfl_xor(mk, m)); }
  float eq = __expf(lq - mq), ek = __expf(lk - mk);
  float sq = eq, sk = ek;
#pragma unroll
  for (int m = 16; m; m >>= 1) { sq += __shfl_xor(sq, m); sk += __shfl_xor(sk, m); }
  float pq = eq / sq, pk = ek / sk;
  atomicAdd(&mq_sum[h * NBK + lane], pq);
  atomicAdd(&mk_sum[h * NBK + lane], pk);
  size_t bh = (size_t)b * HH + h;
  __bf16* qf = Qf + (bh * SS + s) * KFD;
  __bf16* kf = Kf + (bh * SS + s) * KFD;
  qf[2 * lane] = (__bf16)q0; qf[2 * lane + 1] = (__bf16)q1;   // q/8 (folded in Wq)
  kf[2 * lane] = (__bf16)k0; kf[2 * lane + 1] = (__bf16)k1;
  qf[DHD + lane] = (__bf16)(0.1f * pq);                        // CLUSTER_SCORE folded here
  kf[DHD + lane] = (__bf16)pk;
  vT[(bh * DHD + 2 * lane) * SS + s]     = vr[2 * lane];
  vT[(bh * DHD + 2 * lane + 1) * SS + s] = vr[2 * lane + 1];
}

// ---------------- flash attention over packed features (K-dim 96) ----------------
#define KFP  104   // K/Q tile LDS stride
#define PVLD 136   // P and V^T LDS stride
__global__ __launch_bounds__(256) void k_attn(const __bf16* __restrict__ Qf,
                                              const __bf16* __restrict__ Kf,
                                              const __bf16* __restrict__ vT,
                                              __bf16* __restrict__ outp) {
  __shared__ __bf16 sKP[8 * 16 * PVLD];  // union: K tile [128][104]  /  P strips [8][16][136]
  __shared__ __bf16 sVt[64 * PVLD];      // V^T tile [64][128(+pad)]
  const int bh = blockIdx.y;
  const int q0 = blockIdx.x * 128;
  const int t = threadIdx.x, lane = t & 31, w = t >> 5;
  const __bf16* Qb = Qf + ((size_t)bh * SS + q0) * KFD;
  const __bf16* Kb = Kf + (size_t)bh * SS * KFD;
  const __bf16* Vb = vT + (size_t)bh * DHD * SS;

  {  // stage Q tile, then keep fragments in registers
    int row = t >> 1, c = (t & 1) * 48;
    const __bf16* g = Qb + (size_t)row * KFD + c;
    __bf16* l = sKP + row * KFP + c;
#pragma unroll
    for (int i = 0; i < 6; i++) async_cp16(g + 8 * i, l + 8 * i);
    async_wait();
  }
  __syncthreads();
  bf16x16 qfr[3];
#pragma unroll
  for (int kk = 0; kk < 3; kk++) qfr[kk] = load_frag_a(sKP + w * 16 * KFP + kk * 32, KFP);
  __syncthreads();

  f32x8 o[4] = {};
  float mrow[8], lsum[8];
#pragma unroll
  for (int r = 0; r < 8; r++) { mrow[r] = -1e30f; lsum[r] = 0.f; }
  const int lo = lane & 15, hi = lane >> 4;

  for (int kt = 0; kt < SS; kt += 128) {
    {  // stage K feature tile + V^T tile
      int row = t >> 1, c = (t & 1) * 48;
      const __bf16* g = Kb + (size_t)(kt + row) * KFD + c;
      __bf16* l = sKP + row * KFP + c;
      int vrow = t >> 2, vc = (t & 3) * 32;
      const __bf16* gv = Vb + (size_t)vrow * SS + kt + vc;
      __bf16* lv = sVt + vrow * PVLD + vc;
#pragma unroll
      for (int i = 0; i < 6; i++) async_cp16(g + 8 * i, l + 8 * i);
#pragma unroll
      for (int i = 0; i < 4; i++) async_cp16(gv + 8 * i, lv + 8 * i);
      async_wait();
    }
    __syncthreads();

    // S = Qf * Kf^T  (16 rows per wave x 128 cols)
    f32x8 sc[8];
#pragma unroll
    for (int nt = 0; nt < 8; ++nt) {
      f32x8 a = {};
#pragma unroll
      for (int kk = 0; kk < 3; ++kk) {
        bf16x16 bfr = load_frag_b(sKP + (nt * 16) * KFP + kk * 32, KFP);
        a = __builtin_amdgcn_wmma_f32_16x16x32_bf16(false, qfr[kk], false, bfr,
                                                    (short)0, a, false, false);
      }
      sc[nt] = a;
    }
    __syncthreads();  // all waves done with K tile; region becomes P strips

    // online softmax
    float mnew[8], scale[8];
#pragma unroll
    for (int r = 0; r < 8; r++) {
      float mx = sc[0][r];
#pragma unroll
      for (int nt = 1; nt < 8; nt++) mx = fmaxf(mx, sc[nt][r]);
#pragma unroll
      for (int m = 8; m; m >>= 1) mx = fmaxf(mx, __shfl_xor(mx, m));
      mnew[r] = fmaxf(mrow[r], mx);
      scale[r] = __expf(mrow[r] - mnew[r]);
      mrow[r] = mnew[r];
    }
    __bf16* Pst = sKP + w * 16 * PVLD;
#pragma unroll
    for (int r = 0; r < 8; r++) {
      float rs = 0.f;
#pragma unroll
      for (int nt = 0; nt < 8; nt++) {
        float p = __expf(sc[nt][r] - mnew[r]);
        sc[nt][r] = p;
        rs += p;
      }
#pragma unroll
      for (int m = 8; m; m >>= 1) rs += __shfl_xor(rs, m);
      lsum[r] = lsum[r] * scale[r] + rs;
#pragma unroll
      for (int j = 0; j < 4; j++) o[j][r] *= scale[r];
#pragma unroll
      for (int nt = 0; nt < 8; nt++)
        Pst[(r + 8 * hi) * PVLD + nt * 16 + lo] = (__bf16)sc[nt][r];
    }
    // O += P * V   (same-wave DS store->load: hardware keeps DS in order)
#pragma unroll
    for (int kk = 0; kk < 4; ++kk) {
      bf16x16 pa = load_frag_a(Pst + kk * 32, PVLD);
#pragma unroll
      for (int j = 0; j < 4; j++) {
        bf16x16 vb = load_frag_b(sVt + (j * 16) * PVLD + kk * 32, PVLD);
        o[j] = __builtin_amdgcn_wmma_f32_16x16x32_bf16(false, pa, false, vb,
                                                       (short)0, o[j], false, false);
      }
    }
    __syncthreads();  // before next iteration overwrites P / V^T
  }

  int b = bh >> 4, h = bh & 15;
#pragma unroll
  for (int j = 0; j < 4; j++)
#pragma unroll
    for (int r = 0; r < 8; r++) {
      int srow = q0 + w * 16 + r + 8 * hi;
      int f = j * 16 + lo;
      outp[(((size_t)b * SS + srow) * HH + h) * DHD + f] = (__bf16)(o[j][r] / lsum[r]);
    }
}

// ---------------- load-balance loss: 0.5*NB*(mean_h sum_n mq^2 + mean_h sum_n mk^2) ----------------
__global__ __launch_bounds__(512) void k_loss(const float* __restrict__ mq_sum,
                                              const float* __restrict__ mk_sum,
                                              float* __restrict__ out) {
  __shared__ float red[16];
  int t = threadIdx.x;  // 512 = H*NB
  float inv = 1.f / (float)(BB * SS);
  float a = mq_sum[t] * inv, c = mk_sum[t] * inv;
  float val = a * a + c * c;
#pragma unroll
  for (int m = 16; m; m >>= 1) val += __shfl_xor(val, m);
  if ((t & 31) == 0) red[t >> 5] = val;
  __syncthreads();
  if (t < 16) {
    float v = red[t];
#pragma unroll
    for (int m = 8; m; m >>= 1) v += __shfl_xor(v, m);
    if (t == 0) out[(size_t)BSR * DDIM] = 0.5f * (float)NBK * (v / (float)HH);
  }
}

// ---------------- host orchestration ----------------
extern "C" void kernel_launch(void* const* d_in, const int* in_sizes, int n_in,
                              void* d_out, int out_size, void* d_ws, size_t ws_size,
                              hipStream_t stream) {
  (void)in_sizes; (void)n_in; (void)out_size; (void)ws_size;
  const float* x_in  = (const float*)d_in[0];
  const float* ln1_g = (const float*)d_in[1];
  const float* ln1_b = (const float*)d_in[2];
  const float* Wq    = (const float*)d_in[3];
  const float* Wk    = (const float*)d_in[4];
  const float* Wv    = (const float*)d_in[5];
  const float* Whq   = (const float*)d_in[6];
  const float* Whk   = (const float*)d_in[7];
  const float* Wo    = (const float*)d_in[8];
  const float* ln2_g = (const float*)d_in[9];
  const float* ln2_b = (const float*)d_in[10];
  const float* W1    = (const float*)d_in[11];
  const float* b1    = (const float*)d_in[12];
  const float* W2    = (const float*)d_in[13];
  const float* b2    = (const float*)d_in[14];
  float* out = (float*)d_out;

  char* base = (char*)d_ws;
  size_t off = 0;
  auto alloc = [&](size_t bytes) -> void* {
    void* p = base + off;
    off += (bytes + 255) & ~(size_t)255;
    return p;
  };
  __bf16* wq_t = (__bf16*)alloc((size_t)DDIM * DDIM * 2);
  __bf16* wk_t = (__bf16*)alloc((size_t)DDIM * DDIM * 2);
  __bf16* wv_t = (__bf16*)alloc((size_t)DDIM * DDIM * 2);
  __bf16* wo_t = (__bf16*)alloc((size_t)DDIM * DDIM * 2);
  __bf16* w1_t = (__bf16*)alloc((size_t)DDIM * MLPD * 2);
  __bf16* w2_t = (__bf16*)alloc((size_t)MLPD * DDIM * 2);
  __bf16* xhat = (__bf16*)alloc((size_t)BSR * DDIM * 2);
  __bf16* qb   = (__bf16*)alloc((size_t)BSR * DDIM * 2);
  __bf16* kb   = (__bf16*)alloc((size_t)BSR * DDIM * 2);
  __bf16* vb   = (__bf16*)alloc((size_t)BSR * DDIM * 2);
  __bf16* Qf   = (__bf16*)alloc((size_t)BB * HH * SS * KFD * 2);
  __bf16* Kf   = (__bf16*)alloc((size_t)BB * HH * SS * KFD * 2);
  __bf16* vT   = (__bf16*)alloc((size_t)BB * HH * DHD * SS * 2);
  __bf16* aout = (__bf16*)alloc((size_t)BSR * DDIM * 2);
  float*  xres = (float*)alloc((size_t)BSR * DDIM * 4);
  float*  mqs  = (float*)alloc((size_t)2 * HH * NBK * 4);
  float*  mks  = mqs + HH * NBK;
  __bf16* h1   = Qf;  // reuse Qf+Kf+vT region (32MB) after attention

  const dim3 blk(256);
  // weight convert/transpose (Wq carries the 1/sqrt(DH) scale)
  k_convert_t<<<dim3((DDIM * DDIM + 255) / 256), blk, 0, stream>>>(Wq, wq_t, DDIM, DDIM, 0.125f);
  k_convert_t<<<dim3((DDIM * DDIM + 255) / 256), blk, 0, stream>>>(Wk, wk_t, DDIM, DDIM, 1.0f);
  k_convert_t<<<dim3((DDIM * DDIM + 255) / 256), blk, 0, stream>>>(Wv, wv_t, DDIM, DDIM, 1.0f);
  k_convert_t<<<dim3((DDIM * DDIM + 255) / 256), blk, 0, stream>>>(Wo, wo_t, DDIM, DDIM, 1.0f);
  k_convert_t<<<dim3((DDIM * MLPD + 255) / 256), blk, 0, stream>>>(W1, w1_t, DDIM, MLPD, 1.0f);
  k_convert_t<<<dim3((MLPD * DDIM + 255) / 256), blk, 0, stream>>>(W2, w2_t, MLPD, DDIM, 1.0f);
  k_zero<<<dim3(4), blk, 0, stream>>>(mqs, 2 * HH * NBK);

  // LN1 -> QKV projections (bf16 WMMA GEMMs)
  k_layernorm<<<dim3(BSR), blk, 0, stream>>>(x_in, ln1_g, ln1_b, xhat);
  k_gemm<true, false, false, false><<<dim3(DDIM / 128, BSR / 128), blk, 0, stream>>>(
      xhat, wq_t, BSR, DDIM, DDIM, nullptr, nullptr, nullptr, qb);
  k_gemm<true, false, false, false><<<dim3(DDIM / 128, BSR / 128), blk, 0, stream>>>(
      xhat, wk_t, BSR, DDIM, DDIM, nullptr, nullptr, nullptr, kb);
  k_gemm<true, false, false, false><<<dim3(DDIM / 128, BSR / 128), blk, 0, stream>>>(
      xhat, wv_t, BSR, DDIM, DDIM, nullptr, nullptr, nullptr, vb);

  // bucket softmax + feature packing + V transpose
  k_pack<<<dim3(BSR * HH / 8), blk, 0, stream>>>(qb, kb, vb, Whq, Whk, Qf, Kf, vT, mqs, mks);

  // attention
  k_attn<<<dim3(SS / 128, BB * HH), blk, 0, stream>>>(Qf, Kf, vT, aout);

  // output projection + residual
  k_gemm<false, false, false, true><<<dim3(DDIM / 128, BSR / 128), blk, 0, stream>>>(
      aout, wo_t, BSR, DDIM, DDIM, nullptr, x_in, xres, nullptr);

  // LN2 + MLP
  k_layernorm<<<dim3(BSR), blk, 0, stream>>>(xres, ln2_g, ln2_b, xhat);
  k_gemm<true, true, true, false><<<dim3(MLPD / 128, BSR / 128), blk, 0, stream>>>(
      xhat, w1_t, BSR, MLPD, DDIM, b1, nullptr, nullptr, h1);
  k_gemm<false, false, true, true><<<dim3(DDIM / 128, BSR / 128), blk, 0, stream>>>(
      h1, w2_t, BSR, DDIM, MLPD, b2, xres, out, nullptr);

  // scalar aux loss
  k_loss<<<dim3(1), dim3(512), 0, stream>>>(mqs, mks, out);
}
